// EfficientVIMBlock_32246614459066
// MI455X (gfx1250) — matrix operand order repaired and checked by reading, not compiled, not run on hardware
//
#include <hip/hip_runtime.h>
#include <hip/hip_bf16.h>

// ---------------------------------------------------------------------------
// EfficientVIM block for MI455X (gfx1250, wave32, WMMA).
// All large GEMMs use v_wmma_f32_16x16x32_bf16 with fused epilogues
// (bias / GELU / alpha-residual-blend). GEMM tiles are double-buffered in
// LDS; the A tile is staged with GLOBAL_LOAD_ASYNC_TO_LDS_B128 (ASYNCcnt)
// when the builtin is available, overlapped with WMMA compute, plus
// global_prefetch for the K+2 tiles. Bandwidth ops are single-pass
// coalesced kernels with LDS reductions.
//
// Shapes: B=16, C=256, H=W=64 (L=4096), STATE=64, CONV=192, HIDDEN=1024.
// Workspace arena: ~280 MB (lifetime-overlapped regions).
// ---------------------------------------------------------------------------

typedef __attribute__((ext_vector_type(16))) __bf16 v16bf;
typedef __attribute__((ext_vector_type(8)))  float  v8f;
typedef __attribute__((ext_vector_type(4)))  int    i32x4;

#if defined(__AMDGCN__) && __has_builtin(__builtin_amdgcn_global_load_async_to_lds_b128) && __has_builtin(__builtin_amdgcn_s_wait_asynccnt)
#define HAVE_ASYNC_LDS 1
#else
#define HAVE_ASYNC_LDS 0
#endif

#define BATCH 16
#define DIMC  256
#define LLEN  4096
#define STATEC 64
#define CONVC 192
#define HIDC  1024
#define EPSN  1e-5f

// ---------------- workspace offsets (bytes) ----------------
static const size_t OFF_A      = 0;                     //  4 KB  sigmoid(alpha) 4x256
static const size_t OFF_SUMS   = (size_t)16 << 10;      //  2 KB  BN sums (sum,sumsq)
static const size_t OFF_MI     = (size_t)24 << 10;      //  2 KB  BN mean/inv
static const size_t OFF_WBCDT  = (size_t)64 << 10;      // 96 KB  bcdt_w bf16
static const size_t OFF_WF1    = (size_t)256 << 10;     // 512 KB ffn_w1 bf16
static const size_t OFF_WF2    = (size_t)768 << 10;     // 512 KB ffn_w2 bf16
static const size_t OFF_H      = (size_t)2 << 20;       //  1 MB  h   (16,256,64) f32
static const size_t OFF_HZ     = (size_t)3 << 20;       //  2 MB  hz  (16,512,64) f32
static const size_t OFF_G      = (size_t)5 << 20;       //  1 MB  gated (16,256,64) f32
static const size_t OFF_HH2    = (size_t)6 << 20;       // .5 MB  hh2 (16,256,64) bf16
static const size_t OFF_T      = (size_t)8   << 20;     // 64 MB  t0 (K1-K3) / t1 (K11-K13)
static const size_t OFF_X1     = (size_t)72  << 20;     // 64 MB  x1 (K3-K9)
static const size_t OFF_XF     = (size_t)136 << 20;     // 32 MB  xf bf16 (K3-K7) / x3 bf16 (K13-K15)
static const size_t OFF_B1     = (size_t)168 << 20;     // 48 MB  BCdt (K4-K5); x2 f32 64MB (K9-K13)
static const size_t OFF_B2     = (size_t)216 << 20;     // 48 MB  BCdt2 (K5-K6)
static const size_t OFF_HDN    = (size_t)232 << 20;     //  8 MB  hdn bf16 per-batch (K14-K15)
static const size_t OFF_CM     = (size_t)264 << 20;     //  8 MB  Cm bf16 (K5-K9)
static const size_t OFF_ABT    = (size_t)272 << 20;     //  8 MB  (A*B)^T bf16 (K6-K7)

// ---------------- small helpers ----------------
__global__ void zero_f32(float* p, int n) {
  int i = blockIdx.x * 256 + threadIdx.x;
  if (i < n) p[i] = 0.f;
}

__global__ void cvt_bf16(const float* __restrict__ s, __bf16* __restrict__ d, int n) {
  int i = blockIdx.x * 256 + threadIdx.x;
  if (i < n) d[i] = (__bf16)s[i];
}

// ---------------- depthwise 3x3 SAME + optional bias + optional BN stats ----
__global__ __launch_bounds__(256) void dwconv_stats(
    const float* __restrict__ in, const float* __restrict__ w,
    const float* __restrict__ bias, float* __restrict__ outp,
    float* __restrict__ sums, int C) {
  int c = blockIdx.y, b = blockIdx.z, tid = threadIdx.x;
  int oh = blockIdx.x * 4 + (tid >> 6), ow = tid & 63;
  const float* wp  = w + c * 9;
  const float* src = in + ((long)b * C + c) * (long)LLEN;
  float acc = 0.f;
  #pragma unroll
  for (int kh = -1; kh <= 1; ++kh) {
    int ih = oh + kh;
    if (ih < 0 || ih > 63) continue;
    #pragma unroll
    for (int kw = -1; kw <= 1; ++kw) {
      int iw = ow + kw;
      if (iw < 0 || iw > 63) continue;
      acc += wp[(kh + 1) * 3 + (kw + 1)] * src[ih * 64 + iw];
    }
  }
  if (bias) acc += bias[c];
  outp[((long)b * C + c) * (long)LLEN + oh * 64 + ow] = acc;
  if (sums) {
    __shared__ float rs[256], rq[256];
    rs[tid] = acc; rq[tid] = acc * acc;
    __syncthreads();
    for (int o = 128; o; o >>= 1) {
      if (tid < o) { rs[tid] += rs[tid + o]; rq[tid] += rq[tid + o]; }
      __syncthreads();
    }
    if (tid == 0) { atomicAdd(&sums[c], rs[0]); atomicAdd(&sums[256 + c], rq[0]); }
  }
}

// ---------------- finalize BN stats (+sigmoid(alpha) once) ----------------
__global__ void finalize_stats(const float* __restrict__ sums, float* __restrict__ mi,
                               const float* __restrict__ alpha, float* __restrict__ abuf,
                               int do_alpha) {
  int c = threadIdx.x;               // 256 threads, one block
  const float Nf = (float)(BATCH * LLEN);
  float m = sums[c] / Nf;
  float v = sums[256 + c] / Nf - m * m;
  mi[c] = m;
  mi[256 + c] = rsqrtf(v + EPSN);
  if (do_alpha) {
    #pragma unroll
    for (int r = 0; r < 4; ++r)
      abuf[r * 256 + c] = 1.f / (1.f + __expf(-alpha[r * 256 + c]));
  }
}

// ---- residual-blend with BN result + channel LayerNorm -> x1 f32, xf bf16 --
__global__ __launch_bounds__(256) void blend_ln(
    const float* __restrict__ x, const float* __restrict__ t0,
    const float* __restrict__ mi, const float* __restrict__ a0,
    const float* __restrict__ g1, const float* __restrict__ b1,
    const float* __restrict__ lnw, const float* __restrict__ lnb,
    float* __restrict__ x1, __bf16* __restrict__ xf) {
  int b = blockIdx.y;
  int lbase = blockIdx.x * 64;
  int tid = threadIdx.x;
  int li = tid & 63, co = tid >> 6;     // 4 channel-phases x 64 positions
  __shared__ float ps[256], pq[256], mS[64], iS[64];
  float s = 0.f, q = 0.f;
  for (int c = co; c < 256; c += 4) {
    long idx = ((long)b * 256 + c) * (long)LLEN + lbase + li;
    float xv = x[idx], tv = t0[idx];
    float bn = (tv - mi[c]) * mi[256 + c] * g1[c] + b1[c];
    float a = a0[c];
    float v = (1.f - a) * xv + a * bn;
    x1[idx] = v;
    s += v; q += v * v;
  }
  ps[tid] = s; pq[tid] = q;
  __syncthreads();
  if (tid < 64) {
    float ss = ps[tid] + ps[64 + tid] + ps[128 + tid] + ps[192 + tid];
    float qq = pq[tid] + pq[64 + tid] + pq[128 + tid] + pq[192 + tid];
    float m = ss / 256.f;
    float v = qq / 256.f - m * m;
    mS[tid] = m;
    iS[tid] = rsqrtf(v + EPSN);
  }
  __syncthreads();
  for (int c = co; c < 256; c += 4) {
    long idx = ((long)b * 256 + c) * (long)LLEN + lbase + li;
    float v = x1[idx];
    xf[idx] = (__bf16)((v - mS[li]) * iS[li] * lnw[c] + lnb[c]);
  }
}

// ---------------- extract Cm rows as bf16 ----------------
__global__ void cm_extract(const float* __restrict__ bcdt2, __bf16* __restrict__ cm) {
  long i = (long)blockIdx.x * 256 + threadIdx.x;     // 16*64*4096
  long b = i >> 18;
  int  s = (int)((i >> 12) & 63);
  int  l = (int)(i & 4095);
  cm[i] = (__bf16)bcdt2[(b * CONVC + 64 + s) * (long)LLEN + l];
}

// ----- softmax over L of (dt + A[s]); write (Am*Bm)^T bf16 (B,L,64) --------
__global__ __launch_bounds__(256) void softmax_ab(
    const float* __restrict__ bcdt2, const float* __restrict__ Avec,
    __bf16* __restrict__ abt) {
  int s = blockIdx.x, b = blockIdx.y, tid = threadIdx.x;
  const float* dt = bcdt2 + ((long)b * CONVC + 128 + s) * (long)LLEN;
  const float* Bm = bcdt2 + ((long)b * CONVC + s) * (long)LLEN;
  float Aa = Avec[s];
  __shared__ float red[256];
  float mx = -3.0e38f;
  for (int l = tid; l < LLEN; l += 256) mx = fmaxf(mx, dt[l] + Aa);
  red[tid] = mx; __syncthreads();
  for (int o = 128; o; o >>= 1) {
    if (tid < o) red[tid] = fmaxf(red[tid], red[tid + o]);
    __syncthreads();
  }
  mx = red[0]; __syncthreads();
  float sm = 0.f;
  for (int l = tid; l < LLEN; l += 256) sm += __expf(dt[l] + Aa - mx);
  red[tid] = sm; __syncthreads();
  for (int o = 128; o; o >>= 1) {
    if (tid < o) red[tid] += red[tid + o];
    __syncthreads();
  }
  float inv = 1.f / red[0];
  for (int l = tid; l < LLEN; l += 256) {
    float am = __expf(dt[l] + Aa - mx) * inv;
    abt[((long)b * LLEN + l) * 64 + s] = (__bf16)(am * Bm[l]);
  }
}

// ---------------- small dense ops on (B,*,64) state tensors -----------------
__global__ void hz_gemm(const float* __restrict__ w, const float* __restrict__ bias,
                        const float* __restrict__ h, float* __restrict__ hz) {
  long i = (long)blockIdx.x * 256 + threadIdx.x;     // 16*512*64
  long b = i / (512 * 64);
  int  o = (int)((i / 64) % 512);
  int  s = (int)(i & 63);
  const float* hb = h + b * 256 * 64 + s;
  const float* wr = w + o * 256;
  float acc = bias[o];
  for (int c = 0; c < 256; ++c) acc += wr[c] * hb[c * 64];
  hz[i] = acc;
}

__global__ void act_gate(const float* __restrict__ hz, const float* __restrict__ Dp,
                         float* __restrict__ g) {
  long i = (long)blockIdx.x * 256 + threadIdx.x;     // 16*256*64
  long b = i / (256 * 64);
  int  c = (int)((i / 64) % 256);
  int  s = (int)(i & 63);
  float hh = hz[(b * 512 + c) * 64 + s];
  float z  = hz[(b * 512 + 256 + c) * 64 + s];
  float si = 1.f / (1.f + __expf(-z));
  g[i] = hh * (z * si) + hh * Dp[0];
}

__global__ void out_gemm(const float* __restrict__ w, const float* __restrict__ bias,
                         const float* __restrict__ g, __bf16* __restrict__ hh2) {
  long i = (long)blockIdx.x * 256 + threadIdx.x;     // 16*256*64
  long b = i / (256 * 64);
  int  o = (int)((i / 64) % 256);
  int  s = (int)(i & 63);
  const float* gb = g + b * 256 * 64 + s;
  const float* wr = w + o * 256;
  float acc = bias[o];
  for (int c = 0; c < 256; ++c) acc += wr[c] * gb[c * 64];
  hh2[i] = (__bf16)acc;
}

// ---- second BN residual blend -> d_out f32 + x3 bf16 -----------------------
__global__ void blend_bn_store(const float* __restrict__ x2, const float* __restrict__ t1,
                               const float* __restrict__ mi, const float* __restrict__ a2,
                               const float* __restrict__ g2, const float* __restrict__ b2,
                               float* __restrict__ outp, __bf16* __restrict__ xbf) {
  long i = (long)blockIdx.x * 256 + threadIdx.x;     // 16*256*4096
  int c = (int)((i >> 12) & 255);
  float a = a2[c];
  float bn = (t1[i] - mi[c]) * mi[256 + c] * g2[c] + b2[c];
  float v = (1.f - a) * x2[i] + a * bn;
  outp[i] = v;
  xbf[i] = (__bf16)v;
}

// ---------------------------------------------------------------------------
// Generic bf16 WMMA GEMM: C[bz] = epilogue(A[bz] @ B[bz] + bias)
//   A: (M,K) row-major bf16, B: (K,N) row-major bf16.
//   Tile 64x64, BK=32, 256 threads (8 waves); wave (wm,wn) owns 16 rows x
//   32 cols = two 16x16 f32 accumulators of v_wmma_f32_16x16x32_bf16.
//   LDS tiles are double-buffered; A tile staged via async global->LDS
//   (ASYNCcnt) when available; K+2 tiles prefetched into L2.
//   epimode: 0 = +bias, 1 = +bias then GELU(tanh),
//            2 = blend: out = (1-a[m])*res + a[m]*(acc+bias)
// ---------------------------------------------------------------------------
__global__ __launch_bounds__(256) void gemm_bf16_wmma(
    const __bf16* __restrict__ A, const __bf16* __restrict__ B,
    const float* __restrict__ bias, const float* __restrict__ avec,
    const float* __restrict__ res, void* __restrict__ outp,
    int M, int N, int K, long sA, long sB, long sO,
    int epimode, int out_bf16) {
  __shared__ __align__(16) __bf16 As[2][64 * 32];
  __shared__ __align__(16) __bf16 Bs[2][64 * 32];   // stored transposed: Bs[n][k]
  const int bz = blockIdx.z;
  A += (long)bz * sA;
  B += (long)bz * sB;
  const int m0 = blockIdx.y * 64, n0 = blockIdx.x * 64;
  const int tid = threadIdx.x;
  const int lane = tid & 31, wave = tid >> 5;
  const int wm = wave & 3, wn = wave >> 2;
  union FR { unsigned u[8]; v16bf v; };
  v8f acc[2] = {};

  const int ar = tid >> 2, akc = (tid & 3) << 3;    // A staging coords
  const int bk = tid >> 3, bnc = (tid & 7) << 3;    // B staging coords
  const int nk = K >> 5;

  auto stage = [&](int k0, int buf) {
    // A tile (64x32), row-major: one 16B chunk per thread.
    const __bf16* gA = A + (long)(m0 + ar) * K + (k0 + akc);
#if HAVE_ASYNC_LDS
    __builtin_amdgcn_global_load_async_to_lds_b128(
        (__attribute__((address_space(1))) i32x4*)gA,
        (__attribute__((address_space(3))) i32x4*)(&As[buf][ar * 32 + akc]),
        0, 0);
#else
    *(uint4*)(&As[buf][ar * 32 + akc]) = *(const uint4*)gA;
#endif
    // B tile (32x64) -> transposed into LDS (Bs[n][k]) through VGPRs.
    const __bf16* gB = B + (long)(k0 + bk) * N + (n0 + bnc);
    uint4 gv = *(const uint4*)gB;
    const __bf16* e = (const __bf16*)&gv;
    #pragma unroll
    for (int j = 0; j < 8; ++j) Bs[buf][(bnc + j) * 32 + bk] = e[j];
    // prefetch K+2 tiles into GL2 (streamed operands, 23.3 TB/s HBM path)
    __builtin_prefetch(gA + 32, 0, 1);
    __builtin_prefetch(gB + (long)32 * N, 0, 1);
  };

  stage(0, 0);
  for (int ki = 0; ki < nk; ++ki) {
    const int cur = ki & 1;
#if HAVE_ASYNC_LDS
    __builtin_amdgcn_s_wait_asynccnt(0);   // our async A-tile landed in LDS
#endif
    __syncthreads();                        // all waves: tiles[cur] visible,
                                            // buffer cur^1 fully drained
    if (ki + 1 < nk) stage((ki + 1) << 5, cur ^ 1);

    // A fragment: 16x32, lanes 0-15 <-> M rows, lane-half selects K sub-block
    FR af;
    {
      int m = (wm << 4) + (lane & 15), half = lane >> 4;
      #pragma unroll
      for (int j = 0; j < 8; ++j) {
        int kk = ((j >> 2) << 4) + (half << 3) + ((j & 3) << 1);
        af.u[j] = *(const unsigned*)(&As[cur][m * 32 + kk]);
      }
    }
    #pragma unroll
    for (int t = 0; t < 2; ++t) {
      // B fragment: 32x16, lanes <-> N cols, half selects K sub-block (same order as A)
      FR bfv;
      int n = (wn << 5) + (t << 4) + (lane & 15), half = lane >> 4;
      #pragma unroll
      for (int j = 0; j < 8; ++j) {
        int kk = ((j >> 2) << 4) + (half << 3) + ((j & 3) << 1);
        bfv.u[j] = *(const unsigned*)(&Bs[cur][n * 32 + kk]);
      }
      acc[t] = __builtin_amdgcn_wmma_f32_16x16x32_bf16(
          false, af.v, false, bfv.v, (short)0, acc[t], false, false);
    }
  }

  // epilogue: C/D layout -> m = j + 8*(lane>>4), n = lane&15 within 16x16 tile
  #pragma unroll
  for (int t = 0; t < 2; ++t) {
    #pragma unroll
    for (int j = 0; j < 8; ++j) {
      int m = m0 + (wm << 4) + j + ((lane >> 4) << 3);
      int n = n0 + (wn << 5) + (t << 4) + (lane & 15);
      float v = acc[t][j];
      if (bias) v += bias[m];
      if (epimode == 1) {
        float u = 0.7978845608028654f * (v + 0.044715f * v * v * v);
        v = 0.5f * v * (1.f + tanhf(u));
      } else if (epimode == 2) {
        float a = avec[m];
        float r = res[(long)bz * sO + (long)m * N + n];
        v = (1.f - a) * r + a * v;
      }
      long o = (long)bz * sO + (long)m * N + n;
      if (out_bf16) ((__bf16*)outp)[o] = (__bf16)v;
      else          ((float*)outp)[o] = v;
    }
  }
}

// ---------------------------------------------------------------------------
extern "C" void kernel_launch(void* const* d_in, const int* in_sizes, int n_in,
                              void* d_out, int out_size, void* d_ws, size_t ws_size,
                              hipStream_t stream) {
  (void)in_sizes; (void)n_in; (void)out_size; (void)ws_size;
  const float* x      = (const float*)d_in[0];
  const float* alpha  = (const float*)d_in[1];
  const float* dw1_w  = (const float*)d_in[2];
  const float* bn1_g  = (const float*)d_in[3];
  const float* bn1_b  = (const float*)d_in[4];
  const float* dw2_w  = (const float*)d_in[5];
  const float* bn2_g  = (const float*)d_in[6];
  const float* bn2_b  = (const float*)d_in[7];
  const float* ln_w   = (const float*)d_in[8];
  const float* ln_b   = (const float*)d_in[9];
  const float* bcdt_w = (const float*)d_in[10];
  const float* bcdt_b = (const float*)d_in[11];
  const float* mdw_w  = (const float*)d_in[12];
  const float* mdw_b  = (const float*)d_in[13];
  const float* hz_w   = (const float*)d_in[14];
  const float* hz_b   = (const float*)d_in[15];
  const float* out_w  = (const float*)d_in[16];
  const float* out_b  = (const float*)d_in[17];
  const float* Avec   = (const float*)d_in[18];
  const float* Dsc    = (const float*)d_in[19];
  const float* ffn_w1 = (const float*)d_in[20];
  const float* ffn_b1 = (const float*)d_in[21];
  const float* ffn_w2 = (const float*)d_in[22];
  const float* ffn_b2 = (const float*)d_in[23];

  char* wsb = (char*)d_ws;
  float*  abuf  = (float*)(wsb + OFF_A);
  float*  sums  = (float*)(wsb + OFF_SUMS);
  float*  mi    = (float*)(wsb + OFF_MI);
  __bf16* wbcdt = (__bf16*)(wsb + OFF_WBCDT);
  __bf16* wf1   = (__bf16*)(wsb + OFF_WF1);
  __bf16* wf2   = (__bf16*)(wsb + OFF_WF2);
  float*  hbuf  = (float*)(wsb + OFF_H);
  float*  hzbuf = (float*)(wsb + OFF_HZ);
  float*  gbuf  = (float*)(wsb + OFF_G);
  __bf16* hh2   = (__bf16*)(wsb + OFF_HH2);
  float*  tbuf  = (float*)(wsb + OFF_T);     // t0 / t1
  float*  x1    = (float*)(wsb + OFF_X1);
  __bf16* xfbf  = (__bf16*)(wsb + OFF_XF);   // xf / x3 bf16
  float*  bcdt  = (float*)(wsb + OFF_B1);    // BCdt; later x2 f32
  float*  x2    = (float*)(wsb + OFF_B1);
  float*  bcdt2 = (float*)(wsb + OFF_B2);
  __bf16* hdn   = (__bf16*)(wsb + OFF_HDN);
  __bf16* cmbf  = (__bf16*)(wsb + OFF_CM);
  __bf16* abt   = (__bf16*)(wsb + OFF_ABT);
  float*  outF  = (float*)d_out;

  const long N16M = (long)BATCH * DIMC * LLEN;   // 16,777,216

  // --- weight conversion to bf16 ---
  cvt_bf16<<<(CONVC * DIMC + 255) / 256, 256, 0, stream>>>(bcdt_w, wbcdt, CONVC * DIMC);
  cvt_bf16<<<(HIDC * DIMC + 255) / 256, 256, 0, stream>>>(ffn_w1, wf1, HIDC * DIMC);
  cvt_bf16<<<(DIMC * HIDC + 255) / 256, 256, 0, stream>>>(ffn_w2, wf2, DIMC * HIDC);

  // --- stage 1: dwconv1 + BN stats; blend + LayerNorm ---
  zero_f32<<<2, 256, 0, stream>>>(sums, 512);
  dwconv_stats<<<dim3(16, DIMC, BATCH), 256, 0, stream>>>(x, dw1_w, nullptr, tbuf, sums, DIMC);
  finalize_stats<<<1, 256, 0, stream>>>(sums, mi, alpha, abuf, 1);
  blend_ln<<<dim3(64, BATCH), 256, 0, stream>>>(x, tbuf, mi, abuf, bn1_g, bn1_b,
                                                ln_w, ln_b, x1, xfbf);

  // --- mixer: BCdt = bcdt_w @ xf  (WMMA, M=192 N=4096 K=256 per batch) ---
  gemm_bf16_wmma<<<dim3(64, 3, BATCH), 256, 0, stream>>>(
      wbcdt, xfbf, bcdt_b, nullptr, nullptr, bcdt,
      CONVC, LLEN, DIMC, 0L, (long)DIMC * LLEN, (long)CONVC * LLEN, 0, 0);

  // depthwise conv on BCdt (+bias), then Cm extraction + softmax(dt+A)*Bm
  dwconv_stats<<<dim3(16, CONVC, BATCH), 256, 0, stream>>>(bcdt, mdw_w, mdw_b, bcdt2,
                                                           nullptr, CONVC);
  cm_extract<<<(int)(((long)BATCH * 64 * LLEN + 255) / 256), 256, 0, stream>>>(bcdt2, cmbf);
  softmax_ab<<<dim3(STATEC, BATCH), 256, 0, stream>>>(bcdt2, Avec, abt);

  // h = xf @ (Am*Bm)^T   (WMMA, M=256 N=64 K=4096 per batch)
  gemm_bf16_wmma<<<dim3(1, 4, BATCH), 256, 0, stream>>>(
      xfbf, abt, nullptr, nullptr, nullptr, hbuf,
      DIMC, STATEC, LLEN, (long)DIMC * LLEN, (long)LLEN * STATEC,
      (long)DIMC * STATEC, 0, 0);

  // small state-space ops (K=64 outputs; VALU is fine here)
  hz_gemm<<<(int)(((long)BATCH * 512 * 64 + 255) / 256), 256, 0, stream>>>(hz_w, hz_b, hbuf, hzbuf);
  act_gate<<<(int)(((long)BATCH * 256 * 64 + 255) / 256), 256, 0, stream>>>(hzbuf, Dsc, gbuf);
  out_gemm<<<(int)(((long)BATCH * 256 * 64 + 255) / 256), 256, 0, stream>>>(out_w, out_b, gbuf, hh2);

  // y = hh2 @ Cm, fused residual blend: x2 = (1-a1)*x1 + a1*y
  gemm_bf16_wmma<<<dim3(64, 4, BATCH), 256, 0, stream>>>(
      hh2, cmbf, nullptr, abuf + 256, x1, x2,
      DIMC, LLEN, STATEC, (long)DIMC * STATEC, (long)STATEC * LLEN,
      (long)DIMC * LLEN, 2, 0);

  // --- stage 2: dwconv2 + BN; blend -> d_out (x3) + bf16 copy ---
  zero_f32<<<2, 256, 0, stream>>>(sums, 512);
  dwconv_stats<<<dim3(16, DIMC, BATCH), 256, 0, stream>>>(x2, dw2_w, nullptr, tbuf, sums, DIMC);
  finalize_stats<<<1, 256, 0, stream>>>(sums, mi, nullptr, nullptr, 0);
  blend_bn_store<<<(int)((N16M + 255) / 256), 256, 0, stream>>>(x2, tbuf, mi, abuf + 512,
                                                                bn2_g, bn2_b, outF, xfbf);

  // --- FFN per batch: GELU GEMM then blend GEMM in place on d_out ---
  for (int b = 0; b < BATCH; ++b) {
    const __bf16* x3b = xfbf + (long)b * DIMC * LLEN;
    float* ob = outF + (long)b * DIMC * LLEN;
    gemm_bf16_wmma<<<dim3(64, 16, 1), 256, 0, stream>>>(
        wf1, x3b, ffn_b1, nullptr, nullptr, hdn,
        HIDC, LLEN, DIMC, 0L, 0L, 0L, 1, 1);
    gemm_bf16_wmma<<<dim3(64, 4, 1), 256, 0, stream>>>(
        wf2, hdn, ffn_b2, abuf + 768, ob, ob,
        DIMC, LLEN, HIDC, 0L, 0L, 0L, 2, 0);
  }
}
